// RGCNVulClassifier_75084618268945
// MI455X (gfx1250) — matrix operand order repaired and checked by reading, not compiled, not run on hardware
//
#include <hip/hip_runtime.h>
#include <hip/hip_bf16.h>

#define NN   100000   // nodes
#define RR   6        // relations
#define EE   400000   // edges per relation
#define HIDD 128      // F_IN == HID == 128
#define OUTD 2

typedef __attribute__((ext_vector_type(16))) __bf16 v16bf;
typedef __attribute__((ext_vector_type(8)))  float  v8f;

// ---------------------------------------------------------------- degrees
__global__ void degree_kernel(const int* __restrict__ src, const int* __restrict__ dst,
                              float* __restrict__ ns, float* __restrict__ nd) {
    int idx = blockIdx.x * blockDim.x + threadIdx.x;
    if (idx >= RR * EE) return;
    int r = idx / EE;
    atomicAdd(ns + (size_t)r * NN + src[idx], 1.0f);
    atomicAdd(nd + (size_t)r * NN + dst[idx], 1.0f);
}

__global__ void rsqrt_kernel(float* __restrict__ buf, int n) {
    int i = blockIdx.x * blockDim.x + threadIdx.x;
    if (i < n) {
        float v = buf[i];
        buf[i] = (v > 0.0f) ? rsqrtf(v) : 0.0f;
    }
}

// sum biases over relations once (bias is added to every node regardless of degree)
__global__ void bsum_kernel(const float* __restrict__ b1, const float* __restrict__ b2,
                            float* __restrict__ bs1, float* __restrict__ bs2) {
    int j = threadIdx.x;
    if (j < HIDD) {
        float s = 0.0f;
        for (int r = 0; r < RR; ++r) s += b1[r * HIDD + j];
        bs1[j] = s;
    } else {
        int k = j - HIDD;
        float s = 0.0f;
        for (int r = 0; r < RR; ++r) s += b2[r * HIDD + k];
        bs2[k] = s;
    }
}

// ------------------------------------------------- edge gather / scatter-add
// one wave32 per edge; each lane moves 4 contiguous floats (128 = 32*4)
__global__ void __launch_bounds__(256)
gather_kernel(const float* __restrict__ x, const int* __restrict__ src,
              const int* __restrict__ dst, const float* __restrict__ ns,
              float* __restrict__ agg) {
    int e = blockIdx.x * 8 + (threadIdx.x >> 5);
    if (e >= EE) return;
    int lane = threadIdx.x & 31;
    int s = src[e], d = dst[e];
    float c = ns[s];
    const float4 v = *(const float4*)(x + (size_t)s * HIDD + lane * 4);
    float* ap = agg + (size_t)d * HIDD + lane * 4;
    atomicAdd(ap + 0, v.x * c);
    atomicAdd(ap + 1, v.y * c);
    atomicAdd(ap + 2, v.z * c);
    atomicAdd(ap + 3, v.w * c);
}

// ------------------------------------------------- WMMA GEMM:  h += (agg * norm_dst) @ W
// block = 256 threads = 8 waves; block covers 128 rows (M), full 128 cols (N).
// Each wave: one 16-row M-tile, eight 16-col N-tiles, K-loop of 4 x K=32 bf16 WMMA.
__global__ void __launch_bounds__(256)
gemm_wmma_kernel(const float* __restrict__ agg, const float* __restrict__ nd,
                 const float* __restrict__ W, float* __restrict__ hout) {
    __shared__ __bf16 ldsw[HIDD * HIDD];      // 32 KB: W transposed, bf16

    // stage W^T: ldsw[n*128 + k] = W[k*128 + n]
    for (int idx = threadIdx.x; idx < HIDD * HIDD; idx += 256) {
        int k = idx >> 7, n = idx & 127;
        ldsw[n * HIDD + k] = (__bf16)W[idx];
    }
    __syncthreads();

    const int wave  = threadIdx.x >> 5;
    const int lane  = threadIdx.x & 31;
    const int lh    = lane >> 4;              // lane half (K-group select)
    const int ll    = lane & 15;
    const int mbase = blockIdx.x * 128 + wave * 16;

    v8f acc[8];
#pragma unroll
    for (int t = 0; t < 8; ++t)
#pragma unroll
        for (int v = 0; v < 8; ++v) acc[t][v] = 0.0f;

#pragma unroll
    for (int kc = 0; kc < 4; ++kc) {
        // A fragment (16x32 bf16): lane ll = row M; lane half selects K octets.
        // vector elems 0..7  -> K = kc*32 + lh*16 + 0..7
        // vector elems 8..15 -> K = kc*32 + lh*16 + 8..15
        v16bf a;
        int row = mbase + ll;
        if (row < NN) {
            float nr = nd[row];
            const float* ap = agg + (size_t)row * HIDD + kc * 32 + lh * 16;
#pragma unroll
            for (int j = 0; j < 8; ++j) {
                a[j]     = (__bf16)(ap[j]      * nr);
                a[8 + j] = (__bf16)(ap[8 + j] * nr);
            }
        } else {
#pragma unroll
            for (int j = 0; j < 16; ++j) a[j] = (__bf16)0.0f;
        }

#pragma unroll
        for (int nt = 0; nt < 8; ++nt) {
            // B fragment (32x16 bf16): lane ll = col N; lane half = K half;
            // 16 contiguous K values -> one 32B LDS vector load
            const v16bf b = *(const v16bf*)(ldsw + (size_t)(nt * 16 + ll) * HIDD
                                                 + kc * 32 + lh * 16);
            acc[nt] = __builtin_amdgcn_wmma_f32_16x16x32_bf16(
                /*neg_a=*/false, a, /*neg_b=*/false, b,
                /*c_mod=*/(short)0, acc[nt],
                /*reuse_a=*/false, /*reuse_b=*/false);
        }
    }

    // C/D layout: VGPR v -> M = v (lanes 0-15) / v+8 (lanes 16-31); N = ll
#pragma unroll
    for (int nt = 0; nt < 8; ++nt) {
        int col = nt * 16 + ll;
#pragma unroll
        for (int v = 0; v < 8; ++v) {
            int row = mbase + lh * 8 + v;
            if (row < NN) hout[(size_t)row * HIDD + col] += acc[nt][v];
        }
    }
}

// ---------------------------------------------------------------- pointwise
__global__ void relu_bias_kernel(float* __restrict__ h, const float* __restrict__ bs, int n) {
    int i = blockIdx.x * blockDim.x + threadIdx.x;
    if (i < n) h[i] = fmaxf(h[i] + bs[i & (HIDD - 1)], 0.0f);
}

// column partial sums per block, 128 atomics/block
__global__ void pool_kernel(const float* __restrict__ h, float* __restrict__ pool) {
    int j = threadIdx.x;                 // 128 threads = one column each
    int start = blockIdx.x * 256;
    int end = min(start + 256, NN);
    float s = 0.0f;
    for (int i = start; i < end; ++i) s += h[(size_t)i * HIDD + j];
    atomicAdd(&pool[j], s);
}

__global__ void classifier_kernel(const float* __restrict__ pool, const float* __restrict__ bs2,
                                  const float* __restrict__ Wc, const float* __restrict__ bc,
                                  float* __restrict__ out) {
    int o = threadIdx.x;
    if (o < OUTD) {
        float s = bc[o];
        const float invN = 1.0f / (float)NN;
        for (int j = 0; j < HIDD; ++j)
            s += (pool[j] * invN + bs2[j]) * Wc[j * OUTD + o];
        out[o] = s;
    }
}

// ---------------------------------------------------------------- launcher
extern "C" void kernel_launch(void* const* d_in, const int* in_sizes, int n_in,
                              void* d_out, int out_size, void* d_ws, size_t ws_size,
                              hipStream_t stream) {
    const float* x    = (const float*)d_in[0];
    const int*   esrc = (const int*)  d_in[1];
    const int*   edst = (const int*)  d_in[2];
    const float* W1   = (const float*)d_in[3];
    const float* b1   = (const float*)d_in[4];
    const float* W2   = (const float*)d_in[5];
    const float* b2   = (const float*)d_in[6];
    const float* Wc   = (const float*)d_in[7];
    const float* bc   = (const float*)d_in[8];
    float* out = (float*)d_out;

    // workspace layout (floats)
    float* ws = (float*)d_ws;
    size_t off = 0;
    float* ns   = ws + off; off += (size_t)RR * NN;      // norm_src per relation
    float* ndg  = ws + off; off += (size_t)RR * NN;      // norm_dst per relation
    float* agg  = ws + off; off += (size_t)NN * HIDD;    // per-relation aggregation
    float* h1   = ws + off; off += (size_t)NN * HIDD;    // layer-1 output
    float* h2   = ws + off; off += (size_t)NN * HIDD;    // layer-2 output
    float* bs1  = ws + off; off += HIDD;
    float* bs2  = ws + off; off += HIDD;
    float* pool = ws + off; off += HIDD;

    const int gemm_blocks  = (NN + 127) / 128;
    const int edge_blocks  = (EE + 7) / 8;
    const size_t feat_elems = (size_t)NN * HIDD;

    // degrees -> norms (shared by both layers)
    hipMemsetAsync(ns, 0, sizeof(float) * 2 * (size_t)RR * NN, stream); // ns + ndg contiguous
    degree_kernel<<<(RR * EE + 255) / 256, 256, 0, stream>>>(esrc, edst, ns, ndg);
    rsqrt_kernel<<<(2 * RR * NN + 255) / 256, 256, 0, stream>>>(ns, 2 * RR * NN);
    bsum_kernel<<<1, 256, 0, stream>>>(b1, b2, bs1, bs2);

    // ---- layer 1: h1 = relu( sum_r GraphConv_r(x) + sum_r b1_r )
    hipMemsetAsync(h1, 0, sizeof(float) * feat_elems, stream);
    for (int r = 0; r < RR; ++r) {
        hipMemsetAsync(agg, 0, sizeof(float) * feat_elems, stream);
        gather_kernel<<<edge_blocks, 256, 0, stream>>>(
            x, esrc + (size_t)r * EE, edst + (size_t)r * EE, ns + (size_t)r * NN, agg);
        gemm_wmma_kernel<<<gemm_blocks, 256, 0, stream>>>(
            agg, ndg + (size_t)r * NN, W1 + (size_t)r * HIDD * HIDD, h1);
    }
    relu_bias_kernel<<<(int)((feat_elems + 255) / 256), 256, 0, stream>>>(h1, bs1, (int)feat_elems);

    // ---- layer 2: h2 = sum_r GraphConv_r(h1)   (bias folded into pooling)
    hipMemsetAsync(h2, 0, sizeof(float) * feat_elems, stream);
    for (int r = 0; r < RR; ++r) {
        hipMemsetAsync(agg, 0, sizeof(float) * feat_elems, stream);
        gather_kernel<<<edge_blocks, 256, 0, stream>>>(
            h1, esrc + (size_t)r * EE, edst + (size_t)r * EE, ns + (size_t)r * NN, agg);
        gemm_wmma_kernel<<<gemm_blocks, 256, 0, stream>>>(
            agg, ndg + (size_t)r * NN, W2 + (size_t)r * HIDD * HIDD, h2);
    }

    // ---- mean pool + classifier
    hipMemsetAsync(pool, 0, sizeof(float) * HIDD, stream);
    pool_kernel<<<(NN + 255) / 256, 128, 0, stream>>>(h2, pool);
    classifier_kernel<<<1, 32, 0, stream>>>(pool, bs2, Wc, bc, out);
}
